// EdgeConv_5995774345346
// MI455X (gfx1250) — matrix operand order repaired and checked by reading, not compiled, not run on hardware
//
#include <hip/hip_runtime.h>

// ---------------------------------------------------------------------------
// EdgeConv x3 for MI455X (gfx1250, wave32, WMMA)
//   per edge: m = [h[dst], h[src], edge_attr] @ W(144x64) + b  -> scatter-add
//   16-edge tiles -> (16x160) @ (160x64) via v_wmma_f32_16x16x32_bf16
// ---------------------------------------------------------------------------

typedef __bf16 v16bf __attribute__((ext_vector_type(16)));
typedef __bf16 v8bf  __attribute__((ext_vector_type(8)));
typedef __bf16 v4bf  __attribute__((ext_vector_type(4)));
typedef float  v8f   __attribute__((ext_vector_type(8)));
typedef float  v4f   __attribute__((ext_vector_type(4)));
typedef int    v4i   __attribute__((ext_vector_type(4)));

#define DIMF   64      // node feature dim (D == H == O)
#define EDF    16      // edge feature dim
#define K_RAW  144     // 2*DIMF + EDF
#define KP     160     // logical K padded to multiple of 32
#define KSTR   168     // A-row stride in bf16 (84 dwords -> conflict-free banks)
#define KCH    5       // K chunks of 32
#define NT     4       // N tiles of 16 (64 outputs)
#define TILE_E 16      // edges per tile (WMMA M)
#define WPB    8       // waves per block
#define BLOCK  (WPB * 32)

__device__ inline v4bf cvt4(v4f v, int relu) {
    if (relu) {
        v.x = fmaxf(v.x, 0.0f); v.y = fmaxf(v.y, 0.0f);
        v.z = fmaxf(v.z, 0.0f); v.w = fmaxf(v.w, 0.0f);
    }
    v4bf r;
    r[0] = (__bf16)v.x; r[1] = (__bf16)v.y;
    r[2] = (__bf16)v.z; r[3] = (__bf16)v.w;
    return r;
}

__device__ inline void atomAdd(float* p, float v) {
    (void)__hip_atomic_fetch_add(p, v, __ATOMIC_RELAXED, __HIP_MEMORY_SCOPE_AGENT);
}

__global__ __launch_bounds__(BLOCK)
void edge_layer(const float* __restrict__ hin,     // N x 64
                const float* __restrict__ attr,    // E x 16
                const int*   __restrict__ srcIdx,  // E
                const int*   __restrict__ dstIdx,  // E
                const float* __restrict__ W,       // 144 x 64 row-major
                const float* __restrict__ bias,    // 64
                float*       __restrict__ hout,    // N x 64 (atomic accumulate)
                int relu_in, int numTiles, long Etot)
{
    // W in WMMA-B layout: slot (ntile, kch, lane) = 16 contiguous bf16 (20 KB)
    __shared__ __align__(32) __bf16 ldsW[NT * KCH * 32 * 16];
    // per-wave A staging: 16 rows x KSTR bf16 (42 KB total)
    __shared__ __align__(32) __bf16 ldsA[WPB][TILE_E * KSTR];

    const int tid = threadIdx.x;

    // ---- stage W into LDS as bf16 in B-operand layout (zero-pad K>=144) ----
    for (int idx = tid; idx < NT * KCH * 32 * 16; idx += BLOCK) {
        int j     = idx & 15;          // element within lane slot
        int ln    = (idx >> 4) & 31;   // lane the slot belongs to
        int chunk = idx >> 9;          // 0..19
        int kch   = chunk % KCH;
        int ntile = chunk / KCH;
        int n = ntile * 16 + (ln & 15);
        int k = kch * 32 + ((ln >> 4) << 4) + j;   // lanes 16-31 hold K+16..K+31
        float w = (k < K_RAW) ? W[k * DIMF + n] : 0.0f;
        ldsW[idx] = (__bf16)w;
    }
    __syncthreads();

    const int lane = tid & 31;
    // scalarize the wave index -> tile loop + full/tail branch become SALU
    const int wave = __builtin_amdgcn_readfirstlane(tid >> 5);
    const int half = lane >> 4;   // 0 or 1
    const int m    = lane & 15;   // gather row / output column-in-tile
    __bf16* Arow = &ldsA[wave][0];

    // bias per output column (per n-tile), same for all 8 accumulator regs
    float biasv[NT];
    #pragma unroll
    for (int t = 0; t < NT; ++t) biasv[t] = bias[t * 16 + m];

    const long gridWaves = (long)gridDim.x * WPB;
    const long waveId    = (long)blockIdx.x * WPB + wave;

    for (long tile = waveId; tile < numTiles; tile += gridWaves) {
        const long eBase = tile * TILE_E;
        const bool full  = (eBase + TILE_E) <= Etot;   // scalar (uniform) cond

        // ---- gather A tile (16 x 160) into per-wave LDS as bf16 ----
        {
            long e = eBase + m;
            if (!full && e >= Etot) e = 0;        // tail guard (no scatter later)
            const int di = dstIdx[e];
            const int si = srcIdx[e];
            const float* p0 = hin + (long)di * DIMF + half * 32;   // cols 0..63
            const float* p1 = hin + (long)si * DIMF + half * 32;   // cols 64..127
            #pragma unroll
            for (int i = 0; i < 8; ++i) {
                v4f v = *(const v4f*)(p0 + i * 4);
                *(v4bf*)(&Arow[m * KSTR + half * 32 + i * 4]) = cvt4(v, relu_in);
            }
            #pragma unroll
            for (int i = 0; i < 8; ++i) {
                v4f v = *(const v4f*)(p1 + i * 4);
                *(v4bf*)(&Arow[m * KSTR + 64 + half * 32 + i * 4]) = cvt4(v, relu_in);
            }
            if (half == 0) {                      // cols 128..143 = edge_attr
                const float* ap = attr + e * EDF;
                #pragma unroll
                for (int i = 0; i < 4; ++i) {
                    v4f v = *(const v4f*)(ap + i * 4);
                    *(v4bf*)(&Arow[m * KSTR + 128 + i * 4]) = cvt4(v, 0);
                }
            } else {                              // cols 144..159 = zero pad
                v4bf z = {};
                #pragma unroll
                for (int i = 0; i < 4; ++i)
                    *(v4bf*)(&Arow[m * KSTR + 144 + i * 4]) = z;
            }
        }
        // same-wave DS ops are in-order; explicit wait before dependent loads
        asm volatile("s_wait_dscnt 0" ::: "memory");

        // ---- A operands: lane(half) needs K = 16h..16h+15 ----
        v16bf a[KCH];
        #pragma unroll
        for (int kc = 0; kc < KCH; ++kc) {
            v8bf lo = *(const v8bf*)(&Arow[m * KSTR + kc * 32 + half * 16]);
            v8bf hi = *(const v8bf*)(&Arow[m * KSTR + kc * 32 + half * 16 + 8]);
            a[kc] = __builtin_shufflevector(lo, hi,
                        0, 1, 2, 3, 4, 5, 6, 7, 8, 9, 10, 11, 12, 13, 14, 15);
        }

        // ---- 4 independent accumulator chains: kc outer, t inner ----
        v8f acc[NT];
        #pragma unroll
        for (int t = 0; t < NT; ++t) {
            #pragma unroll
            for (int i = 0; i < 8; ++i) acc[t][i] = biasv[t];  // +bias per edge
        }
        #pragma unroll
        for (int kc = 0; kc < KCH; ++kc) {
            #pragma unroll
            for (int t = 0; t < NT; ++t) {
                v16bf bm = *(const v16bf*)(&ldsW[((t * KCH + kc) * 32 + lane) * 16]);
                acc[t] = __builtin_amdgcn_wmma_f32_16x16x32_bf16(
                             false, a[kc], false, bm, (short)0, acc[t], false, false);
            }
        }

        // ---- scatter-add: row M = i + 8*half, col N = t*16 + m ----
        if (full) {
            // 8 consecutive dst indices per lane -> two b128 loads
            const int* dp = dstIdx + eBase + half * 8;
            v4i d0 = *(const v4i*)(dp);
            v4i d1 = *(const v4i*)(dp + 4);
            float* rowp[8];
            rowp[0] = hout + (long)d0.x * DIMF;  rowp[1] = hout + (long)d0.y * DIMF;
            rowp[2] = hout + (long)d0.z * DIMF;  rowp[3] = hout + (long)d0.w * DIMF;
            rowp[4] = hout + (long)d1.x * DIMF;  rowp[5] = hout + (long)d1.y * DIMF;
            rowp[6] = hout + (long)d1.z * DIMF;  rowp[7] = hout + (long)d1.w * DIMF;
            #pragma unroll
            for (int t = 0; t < NT; ++t) {
                const int col = t * 16 + m;
                #pragma unroll
                for (int i = 0; i < 8; ++i)
                    atomAdd(rowp[i] + col, acc[t][i]);
            }
        } else {
            #pragma unroll
            for (int i = 0; i < 8; ++i) {
                long e = eBase + i + half * 8;
                if (e < Etot) {
                    float* rp = hout + (long)dstIdx[e] * DIMF;
                    #pragma unroll
                    for (int t = 0; t < NT; ++t)
                        atomAdd(rp + t * 16 + m, acc[t][i]);
                }
            }
        }
    }
}

__global__ __launch_bounds__(256)
void zero_f32(float* __restrict__ p, long n) {
    long i      = (long)blockIdx.x * blockDim.x + threadIdx.x;
    long stride = (long)gridDim.x * blockDim.x;
    for (; i < n; i += stride) p[i] = 0.0f;
}

extern "C" void kernel_launch(void* const* d_in, const int* in_sizes, int n_in,
                              void* d_out, int out_size, void* d_ws, size_t ws_size,
                              hipStream_t stream) {
    const float* x    = (const float*)d_in[0];
    const float* attr = (const float*)d_in[1];
    const int*   eidx = (const int*)  d_in[2];
    const float* W1   = (const float*)d_in[3];
    const float* b1   = (const float*)d_in[4];
    const float* W2   = (const float*)d_in[5];
    const float* b2   = (const float*)d_in[6];
    const float* W3   = (const float*)d_in[7];
    const float* b3   = (const float*)d_in[8];
    float* out = (float*)d_out;

    const long N = in_sizes[0] / DIMF;
    const long E = in_sizes[2] / 2;
    const int  numTiles = (int)((E + TILE_E - 1) / TILE_E);
    const int* srcIdx = eidx;        // edge_index[0] = source
    const int* dstIdx = eidx + E;    // edge_index[1] = target

    float* h1 = (float*)d_ws;        // N x 64
    float* h2 = h1 + N * DIMF;       // N x 64
    const long NH = N * DIMF;

    dim3 zg(1024), zb(256);
    dim3 eg(1024), eb(BLOCK);

    // layer 1: in = x (no relu on gather), out = h1
    zero_f32<<<zg, zb, 0, stream>>>(h1, NH);
    edge_layer<<<eg, eb, 0, stream>>>(x,  attr, srcIdx, dstIdx, W1, b1, h1, 0, numTiles, E);
    // layer 2: in = relu(h1) fused into gather, out = h2
    zero_f32<<<zg, zb, 0, stream>>>(h2, NH);
    edge_layer<<<eg, eb, 0, stream>>>(h1, attr, srcIdx, dstIdx, W2, b2, h2, 1, numTiles, E);
    // layer 3: in = relu(h2) fused into gather, out = d_out (no final relu)
    zero_f32<<<zg, zb, 0, stream>>>(out, NH);
    edge_layer<<<eg, eb, 0, stream>>>(h2, attr, srcIdx, dstIdx, W3, b3, out, 1, numTiles, E);
}
